// MiniQwenAttention_15229954031827
// MI455X (gfx1250) — matrix operand order repaired and checked by reading, not compiled, not run on hardware
//
#include <hip/hip_runtime.h>
#include <cstdint>

#define HIDDEN 2048
#define NHEADS 16
#define NKV 4
#define HDIM 128
#define SEQ 2048
#define BATCH 2
#define MTOT (BATCH * SEQ)   // 4096 rows
#define KVDIM (NKV * HDIM)   // 512

typedef __attribute__((ext_vector_type(16))) __bf16          v16bf;
typedef __attribute__((ext_vector_type(8)))  float           v8f;
typedef __attribute__((ext_vector_type(8)))  unsigned short  v8u16;

union FragU { v8u16 h[2]; v16bf v; };

__device__ __forceinline__ unsigned short f2bf(float f) {
  union { __bf16 h; unsigned short u; } c;
  c.h = (__bf16)f;                       // native f32->bf16 cvt path
  return c.u;
}
__device__ __forceinline__ float bf2f(unsigned short h) {
  union { uint32_t u; float f; } c; c.u = ((uint32_t)h) << 16;
  return c.f;
}
__device__ __forceinline__ v8f vzero8() {
  v8f z;
#pragma unroll
  for (int i = 0; i < 8; ++i) z[i] = 0.f;
  return z;
}
__device__ __forceinline__ v8f wmma_bf16(v16bf a, v16bf b, v8f c) {
  // D(16x16,f32) = A(16x32,bf16) * B(32x16,bf16) + C
  return __builtin_amdgcn_wmma_f32_16x16x32_bf16(false, a, false, b, (short)0, c,
                                                 false, false);
}

// CDNA5 async memory->LDS copy, 16B per lane, tracked by ASYNCcnt.
// VDST = per-lane LDS byte address; generic pointers to __shared__ carry the
// LDS offset in their low 32 bits (ISA 10.2 aperture rules).
__device__ __forceinline__ void async_copy16(void* lds_ptr, const void* gptr) {
  unsigned int       loff = (unsigned int)(uintptr_t)lds_ptr;
  unsigned long long ga   = (unsigned long long)(uintptr_t)gptr;
  asm volatile("global_load_async_to_lds_b128 %0, %1, off"
               :: "v"(loff), "v"(ga) : "memory");
}
__device__ __forceinline__ void wait_async0() {
  asm volatile("s_wait_asynccnt 0" ::: "memory");
}

// ---------------------------------------------------------------------------
// GEMM: C[M,N] = A[M,K] * W[N,K]^T (+ bias). A is fp32 or bf16 (template),
// W/bias fp32. Output fp32 or bf16 (template). Block tile 128(M) x 64(N),
// 8 waves (wave32), 32x32 per wave, K-step 32 via WMMA bf16.
// ---------------------------------------------------------------------------
template <int A_BF16, int OUT_BF16>
__global__ __launch_bounds__(256) void gemm_wmma(const void* __restrict__ Aptr,
                                                 const float* __restrict__ W,
                                                 const float* __restrict__ bias,
                                                 void* __restrict__ Cptr,
                                                 int M, int N, int K) {
  (void)M;
  __shared__ __align__(16) unsigned short As[128 * 48]; // [row][k], padded
  __shared__ __align__(16) unsigned short Bs[32 * 80];  // [k][n],  transposed W

  const int tid  = threadIdx.x;
  const int lane = tid & 31;
  const int w    = tid >> 5;
  const int wm   = w >> 1;  // 0..3
  const int wn   = w & 1;   // 0..1
  const int blockN = blockIdx.x * 64;
  const int blockM = blockIdx.y * 128;

  const float*          Af = (const float*)Aptr;
  const unsigned short* Au = (const unsigned short*)Aptr;

  v8f acc[2][2];
#pragma unroll
  for (int i = 0; i < 2; ++i)
#pragma unroll
    for (int j = 0; j < 2; ++j) acc[i][j] = vzero8();

  for (int k0 = 0; k0 < K; k0 += 32) {
    __syncthreads();
    // ---- stage A tile (128 x 32) -> bf16 LDS ----
    if (A_BF16) {
      // pure 16B copies: use CDNA5 async global->LDS path (ASYNCcnt)
      int row = tid >> 2;              // 0..63, two passes
      int c8  = (tid & 3) * 8;
#pragma unroll
      for (int p = 0; p < 2; ++p) {
        int r = row + p * 64;
        async_copy16(&As[r * 48 + c8],
                     Au + (size_t)(blockM + r) * K + k0 + c8);
      }
    } else {
      int row = tid >> 3;              // 0..31, four passes
      int c4  = (tid & 7) * 4;
#pragma unroll
      for (int p = 0; p < 4; ++p) {
        int r = row + p * 32;
        float4 x = *(const float4*)(Af + (size_t)(blockM + r) * K + k0 + c4);
        ushort4 hh;
        hh.x = f2bf(x.x); hh.y = f2bf(x.y); hh.z = f2bf(x.z); hh.w = f2bf(x.w);
        *(ushort4*)(&As[r * 48 + c4]) = hh;
      }
    }
    // ---- stage W tile (64 x 32) transposed -> Bs[k][n] ----
    {
      int n  = tid >> 3;               // 0..31, two passes
      int c4 = (tid & 7) * 4;
#pragma unroll
      for (int p = 0; p < 2; ++p) {
        int nn = n + p * 32;
        float4 x = *(const float4*)(W + (size_t)(blockN + nn) * K + k0 + c4);
        Bs[(c4 + 0) * 80 + nn] = f2bf(x.x);
        Bs[(c4 + 1) * 80 + nn] = f2bf(x.y);
        Bs[(c4 + 2) * 80 + nn] = f2bf(x.z);
        Bs[(c4 + 3) * 80 + nn] = f2bf(x.w);
      }
    }
    if (k0 + 32 < K) { // prefetch next W tile -> global_prefetch_b8
      __builtin_prefetch(W + (size_t)(blockN + (tid & 63)) * K + k0 + 32, 0, 1);
    }
    if (A_BF16) wait_async0();   // barrier does not cover ASYNCcnt
    __syncthreads();

    // ---- fragments + 4 WMMA ----
    FragU a0, a1, b0, b1;
    const int kb  = (lane >> 4) * 8;                 // A halves: K 0-7 / 8-15
    const int ar0 = wm * 32 + (lane & 15);
    const int ar1 = ar0 + 16;
    a0.h[0] = *(const v8u16*)(&As[ar0 * 48 + kb]);
    a0.h[1] = *(const v8u16*)(&As[ar0 * 48 + kb + 16]);
    a1.h[0] = *(const v8u16*)(&As[ar1 * 48 + kb]);
    a1.h[1] = *(const v8u16*)(&As[ar1 * 48 + kb + 16]);
    const int nb0 = wn * 32 + (lane & 15);
#pragma unroll
    for (int s = 0; s < 16; ++s) {
      const int kk = kb + (s < 8 ? s : s + 8);
      b0.h[s >> 3][s & 7] = Bs[kk * 80 + nb0];
      b1.h[s >> 3][s & 7] = Bs[kk * 80 + nb0 + 16];
    }

    acc[0][0] = wmma_bf16(a0.v, b0.v, acc[0][0]);
    acc[0][1] = wmma_bf16(a0.v, b1.v, acc[0][1]);
    acc[1][0] = wmma_bf16(a1.v, b0.v, acc[1][0]);
    acc[1][1] = wmma_bf16(a1.v, b1.v, acc[1][1]);
  }

  // ---- epilogue: C layout lane=(N), vgpr elem=(M row) ----
  float*          Cf = (float*)Cptr;
  unsigned short* Cu = (unsigned short*)Cptr;
#pragma unroll
  for (int tm = 0; tm < 2; ++tm) {
    const int rowBase = blockM + wm * 32 + tm * 16 + ((lane >> 4) << 3);
#pragma unroll
    for (int tn = 0; tn < 2; ++tn) {
      const int col = blockN + wn * 32 + tn * 16 + (lane & 15);
      const float bv = bias ? bias[col] : 0.f;
#pragma unroll
      for (int r = 0; r < 8; ++r) {
        const float  val = acc[tm][tn][r] + bv;
        const size_t off = (size_t)(rowBase + r) * N + col;
        if (OUT_BF16) Cu[off] = f2bf(val);
        else          Cf[off] = val;
      }
    }
  }
}

// ---------------------------------------------------------------------------
// RoPE, in place on bf16 [rows, cols] with head_dim 128 (rotate pairs d,d+64)
// ---------------------------------------------------------------------------
__global__ __launch_bounds__(256) void rope_kernel(unsigned short* __restrict__ x,
                                                   const int* __restrict__ pos_ids,
                                                   const float* __restrict__ cosT,
                                                   const float* __restrict__ sinT,
                                                   int rows, int cols) {
  const int perRow = cols >> 1;
  const int idx = blockIdx.x * blockDim.x + threadIdx.x;
  if (idx >= rows * perRow) return;
  const int row = idx / perRow;
  const int j   = idx - row * perRow;
  const int h   = j >> 6;       // head
  const int d   = j & 63;       // rotation index
  const int s   = row % SEQ;
  const int p   = pos_ids[s];
  const float c  = cosT[p * HDIM + d];
  const float sn = sinT[p * HDIM + d];
  unsigned short* base = x + (size_t)row * cols + h * HDIM;
  const float x1 = bf2f(base[d]);
  const float x2 = bf2f(base[d + 64]);
  base[d]      = f2bf(x1 * c - x2 * sn);
  base[d + 64] = f2bf(x2 * c + x1 * sn);
}

// ---------------------------------------------------------------------------
// Flash attention (non-causal, additive mask over keys), GQA 16Q/4KV heads.
// Block = 4 waves, 64 q-rows per block (16 per wave), K-tiles of 32.
// ---------------------------------------------------------------------------
__global__ __launch_bounds__(128) void attn_wmma(const unsigned short* __restrict__ Qbf,
                                                 const unsigned short* __restrict__ Kbf,
                                                 const unsigned short* __restrict__ Vbf,
                                                 const float* __restrict__ mask,
                                                 unsigned short* __restrict__ Obf) {
  __shared__ __align__(16) unsigned short Kt[128 * 48];   // K^T: [d][kcol]
  __shared__ __align__(16) unsigned short Vt[32 * 136];   // V  : [krow][d]
  __shared__ __align__(16) unsigned short Pw[4][16 * 48]; // per-wave P scratch

  const int tid  = threadIdx.x;
  const int lane = tid & 31;
  const int w    = tid >> 5;        // 0..3
  const int b    = blockIdx.z;
  const int h    = blockIdx.y;
  const int kvh  = h >> 2;          // GQA group of 4
  const int qrow0 = b * SEQ + blockIdx.x * 64 + w * 16;
  const int hq = h * HDIM;
  const int hk = kvh * HDIM;

  // Q fragments: 16 rows x 128 d as 4 A-fragments (K chunks of 32)
  FragU qa[4];
  {
    const unsigned short* qp =
        Qbf + (size_t)(qrow0 + (lane & 15)) * HIDDEN + hq + ((lane >> 4) << 3);
#pragma unroll
    for (int c = 0; c < 4; ++c) {
      qa[c].h[0] = *(const v8u16*)(qp + c * 32);
      qa[c].h[1] = *(const v8u16*)(qp + c * 32 + 16);
    }
  }

  v8f o[8];
#pragma unroll
  for (int i = 0; i < 8; ++i) o[i] = vzero8();
  float mrow[8], lsum[8];
#pragma unroll
  for (int r = 0; r < 8; ++r) { mrow[r] = -1e30f; lsum[r] = 0.f; }
  const float scale = 0.08838834764831845f; // 1/sqrt(128)

  const int kr  = tid & 31;
  const int dc0 = tid >> 5;
  const int bk  = (lane & 15) + ((lane >> 4) << 4); // B fragment K-row

  for (int kt = 0; kt < SEQ / 32; ++kt) {
    __syncthreads();
    // ---- stage K tile transposed (via VGPR, needs transpose) and
    //      V tile natural (pure 16B copy -> CDNA5 async global->LDS) ----
    {
      const size_t krow = (size_t)(b * SEQ + kt * 32 + kr) * KVDIM + hk;
#pragma unroll
      for (int p = 0; p < 4; ++p) {
        const int d0 = (dc0 + p * 4) * 8;
        v8u16 kx = *(const v8u16*)(Kbf + krow + d0);
#pragma unroll
        for (int j = 0; j < 8; ++j) Kt[(d0 + j) * 48 + kr] = kx[j];
        async_copy16(&Vt[kr * 136 + d0], Vbf + krow + d0);
      }
    }
    wait_async0();
    __syncthreads();

    // ---- scores: S = Q * K^T for 32 key columns (2 x 16) ----
    v8f c0 = vzero8(), c1 = vzero8();
#pragma unroll
    for (int c = 0; c < 4; ++c) {
      FragU bb0, bb1;
#pragma unroll
      for (int s = 0; s < 16; ++s) {
        const int kd = c * 32 + ((lane >> 4) << 3) + (s < 8 ? s : s + 8);
        bb0.h[s >> 3][s & 7] = Kt[kd * 48 + (lane & 15)];
        bb1.h[s >> 3][s & 7] = Kt[kd * 48 + 16 + (lane & 15)];
      }
      c0 = wmma_bf16(qa[c].v, bb0.v, c0);
      c1 = wmma_bf16(qa[c].v, bb1.v, c1);
    }

    // ---- online softmax (rows per VGPR element, key-cols per lane-half) ----
    const float* mp = mask + (size_t)b * SEQ + kt * 32;
    const float m0 = mp[lane & 15];
    const float m1 = mp[16 + (lane & 15)];
    float s0[8], s1[8], mnew[8], alpha[8], p0[8], p1[8];
#pragma unroll
    for (int r = 0; r < 8; ++r) {
      s0[r] = c0[r] * scale + m0;
      s1[r] = c1[r] * scale + m1;
      float t = fmaxf(s0[r], s1[r]);
      t = fmaxf(t, __shfl_xor(t, 1, 32));
      t = fmaxf(t, __shfl_xor(t, 2, 32));
      t = fmaxf(t, __shfl_xor(t, 4, 32));
      t = fmaxf(t, __shfl_xor(t, 8, 32));
      mnew[r]  = fmaxf(mrow[r], t);
      alpha[r] = __expf(mrow[r] - mnew[r]);
      p0[r] = __expf(s0[r] - mnew[r]);
      p1[r] = __expf(s1[r] - mnew[r]);
      float rs = p0[r] + p1[r];
      rs += __shfl_xor(rs, 1, 32);
      rs += __shfl_xor(rs, 2, 32);
      rs += __shfl_xor(rs, 4, 32);
      rs += __shfl_xor(rs, 8, 32);
      lsum[r] = lsum[r] * alpha[r] + rs;
      mrow[r] = mnew[r];
    }
#pragma unroll
    for (int dc = 0; dc < 8; ++dc)
#pragma unroll
      for (int r = 0; r < 8; ++r) o[dc][r] *= alpha[r];

    // ---- P: C-layout -> A-layout via per-wave LDS scratch ----
    unsigned short* P = &Pw[w][0];
    {
      const int n  = lane & 15;
      const int rh = (lane >> 4) * 8;
#pragma unroll
      for (int r = 0; r < 8; ++r) {
        P[(rh + r) * 48 + n]      = f2bf(p0[r]);
        P[(rh + r) * 48 + 16 + n] = f2bf(p1[r]);
      }
    }
    asm volatile("s_wait_dscnt 0" ::: "memory");
    FragU pa;
    {
      const int row = lane & 15;
      const int kb  = (lane >> 4) * 8;
      pa.h[0] = *(const v8u16*)(&P[row * 48 + kb]);
      pa.h[1] = *(const v8u16*)(&P[row * 48 + kb + 16]);
    }

    // ---- O += P * V ----
#pragma unroll
    for (int dc = 0; dc < 8; ++dc) {
      FragU bv;
#pragma unroll
      for (int s = 0; s < 16; ++s) {
        const int kk = ((lane >> 4) << 3) + (s < 8 ? s : s + 8);
        bv.h[s >> 3][s & 7] = Vt[kk * 136 + dc * 16 + (lane & 15)];
      }
      o[dc] = wmma_bf16(pa.v, bv.v, o[dc]);
    }
  }

  // ---- normalize + store bf16 ----
  {
    const int n  = lane & 15;
    const int rh = (lane >> 4) * 8;
#pragma unroll
    for (int r = 0; r < 8; ++r) {
      const float inv = 1.f / lsum[r];
      const size_t row = (size_t)(qrow0 + rh + r);
#pragma unroll
      for (int dc = 0; dc < 8; ++dc)
        Obf[row * HIDDEN + hq + dc * 16 + n] = f2bf(o[dc][r] * inv);
    }
  }
}

// ---------------------------------------------------------------------------
extern "C" void kernel_launch(void* const* d_in, const int* in_sizes, int n_in,
                              void* d_out, int out_size, void* d_ws, size_t ws_size,
                              hipStream_t stream) {
  (void)in_sizes; (void)n_in; (void)out_size; (void)ws_size;
  const float* X    = (const float*)d_in[0];
  const float* mask = (const float*)d_in[1];
  const int*   pos  = (const int*)d_in[2];
  const float* qw   = (const float*)d_in[3];
  const float* qb   = (const float*)d_in[4];
  const float* kw   = (const float*)d_in[5];
  const float* kb   = (const float*)d_in[6];
  const float* vw   = (const float*)d_in[7];
  const float* vb   = (const float*)d_in[8];
  const float* ow   = (const float*)d_in[9];
  const float* cosT = (const float*)d_in[10];
  const float* sinT = (const float*)d_in[11];

  unsigned short* Qbf = (unsigned short*)d_ws;              // 4096x2048 bf16
  unsigned short* Kbf = Qbf + (size_t)MTOT * HIDDEN;        // 4096x512
  unsigned short* Vbf = Kbf + (size_t)MTOT * KVDIM;         // 4096x512
  unsigned short* Abf = Vbf + (size_t)MTOT * KVDIM;         // 4096x2048

  const dim3 blk(256);
  // Q/K/V projections (fp32 in, bf16 out, fused bias)
  gemm_wmma<0, 1><<<dim3(HIDDEN / 64, MTOT / 128), blk, 0, stream>>>(
      X, qw, qb, Qbf, MTOT, HIDDEN, HIDDEN);
  gemm_wmma<0, 1><<<dim3(KVDIM / 64, MTOT / 128), blk, 0, stream>>>(
      X, kw, kb, Kbf, MTOT, KVDIM, HIDDEN);
  gemm_wmma<0, 1><<<dim3(KVDIM / 64, MTOT / 128), blk, 0, stream>>>(
      X, vw, vb, Vbf, MTOT, KVDIM, HIDDEN);

  // RoPE on Q and K (in place)
  rope_kernel<<<(MTOT * (HIDDEN / 2) + 255) / 256, 256, 0, stream>>>(
      Qbf, pos, cosT, sinT, MTOT, HIDDEN);
  rope_kernel<<<(MTOT * (KVDIM / 2) + 255) / 256, 256, 0, stream>>>(
      Kbf, pos, cosT, sinT, MTOT, KVDIM);

  // Flash attention
  attn_wmma<<<dim3(SEQ / 64, NHEADS, BATCH), dim3(128), 0, stream>>>(
      Qbf, Kbf, Vbf, mask, Abf);

  // Output projection (bf16 in, fp32 out, no bias) -> d_out
  gemm_wmma<1, 0><<<dim3(HIDDEN / 64, MTOT / 128), blk, 0, stream>>>(
      Abf, ow, nullptr, d_out, MTOT, HIDDEN, HIDDEN);
}
